// CNN_RNN_30416958390487
// MI455X (gfx1250) — matrix operand order repaired and checked by reading, not compiled, not run on hardware
//
#include <hip/hip_runtime.h>
#include <cstdint>
#include <cstddef>

// ---------------------------------------------------------------------------
// CNN + GRU forward for MI455X (gfx1250, wave32, WMMA).
//
// Perf model (MI455X): GRU weight streaming dominates. w_hh is re-read each of
// the 20 sequential GRU steps (M=8 -> ~16 FLOP/weight, hopelessly memory
// bound; 236-472MB > 192MB L2). bf16 weights + v_wmma_f32_16x16x32_bf16
// (f32 accumulate) halve HBM bytes: ~6.6GB total -> ~0.3ms at 23.3TB/s.
// gi is computed as ONE GEMM over all 20 timesteps (M=160) for weight reuse.
// GEMM inner loop is software double-buffered (kchunk % 64 == 0) so fragment
// loads for K+32 overlap the WMMAs on K (partial s_wait_loadcnt).
// K and the C stride are compile-time constants so all MT/NT row addresses are
// instruction-immediate offsets from 2 base pointers (no VGPR spills).
// Cache hints: w_ih stream (read once) -> non-temporal loads; w_hh stream
// (re-read 20x, > L2) -> regular loads + global_prefetch_b8 at distance.
// ---------------------------------------------------------------------------

typedef __bf16 bf16;
typedef __attribute__((ext_vector_type(16))) __bf16 v16bf;
typedef __attribute__((ext_vector_type(8)))  __bf16 v8bf;
typedef __attribute__((ext_vector_type(8)))  float  v8f;

#define HID   6272      // 16*28*14
#define N3    18816     // 3*HID
#define NWIN  20
#define NBAT  160       // 8*NWIN

__device__ __forceinline__ unsigned short f2bf_u16(float f) {
  unsigned u = __builtin_bit_cast(unsigned, f);
  u += 0x7FFFu + ((u >> 16) & 1u);          // round-to-nearest-even
  return (unsigned short)(u >> 16);
}

// ---- fused window extract + conv1(9x9,pad2) + leaky + maxpool3 -------------
// out: (160, 8, 84, 41) f32 ; input x: (8, 257, 1345)
__global__ void k_conv1_pool(const float* __restrict__ x, const float* __restrict__ w,
                             const float* __restrict__ bias, float* __restrict__ out) {
  int idx = blockIdx.x * blockDim.x + threadIdx.x;
  if (idx >= NBAT * 8 * 84 * 41) return;
  int xo = idx % 41; int tmp = idx / 41;
  int yo = tmp % 84; tmp /= 84;
  int oc = tmp % 8;  int nb = tmp / 8;          // nb = b*20 + t
  int b = nb / NWIN, t = nb % NWIN;
  // win[nb][0][u][v] = x[b][u+1][t*64+v], u in [0,255], v in [0,127]
  const float* xb = x + (size_t)b * 257 * 1345 + 1345 + (size_t)t * 64;
  float wreg[81];
#pragma unroll
  for (int q = 0; q < 81; ++q) wreg[q] = w[oc * 81 + q];
  float bb = bias[oc];
  float best = -3.4e38f;
  for (int py = 0; py < 3; ++py)
    for (int px = 0; px < 3; ++px) {
      int cy = yo * 3 + py, cx = xo * 3 + px;   // conv coords: 252 x 124
      float s = bb;
#pragma unroll
      for (int i = 0; i < 9; ++i) {
        int u = cy - 2 + i;
        if (u < 0 || u > 255) continue;
#pragma unroll
        for (int j = 0; j < 9; ++j) {
          int v = cx - 2 + j;
          if (v < 0 || v > 127) continue;
          s += xb[u * 1345 + v] * wreg[i * 9 + j];
        }
      }
      s = (s > 0.0f) ? s : 0.01f * s;
      best = fmaxf(best, s);
    }
  out[idx] = best;
}

// ---- conv2(4x4,pad2) + leaky + maxpool3, writes feats in GEMM row order ----
// featsA: (160, 6272) f32, row = t*8 + b, col = (oc*28+yo)*14+xo
__global__ void k_conv2_pool(const float* __restrict__ c1, const float* __restrict__ w,
                             const float* __restrict__ bias, float* __restrict__ featsA) {
  int idx = blockIdx.x * blockDim.x + threadIdx.x;
  if (idx >= NBAT * 16 * 28 * 14) return;
  int xo = idx % 14; int tmp = idx / 14;
  int yo = tmp % 28; tmp /= 28;
  int oc = tmp % 16; int nb = tmp / 16;
  int b = nb / NWIN, t = nb % NWIN;
  float wreg[128];
#pragma unroll
  for (int q = 0; q < 128; ++q) wreg[q] = w[oc * 128 + q];
  float bb = bias[oc];
  const float* cin = c1 + (size_t)nb * 8 * 84 * 41;
  float best = -3.4e38f;
  for (int py = 0; py < 3; ++py)
    for (int px = 0; px < 3; ++px) {
      int cy = yo * 3 + py, cx = xo * 3 + px;   // conv coords: 85 x 42
      float s = bb;
#pragma unroll
      for (int ic = 0; ic < 8; ++ic)
#pragma unroll
        for (int i = 0; i < 4; ++i) {
          int u = cy - 2 + i;
          if (u < 0 || u > 83) continue;
#pragma unroll
          for (int j = 0; j < 4; ++j) {
            int v = cx - 2 + j;
            if (v < 0 || v > 40) continue;
            s += cin[((size_t)ic * 84 + u) * 41 + v] * wreg[ic * 16 + i * 4 + j];
          }
        }
      s = (s > 0.0f) ? s : 0.01f * s;
      best = fmaxf(best, s);
    }
  featsA[(size_t)(t * 8 + b) * HID + (oc * 28 + yo) * 14 + xo] = best;
}

// ---- fp32 -> bf16 packed converter (4 elems/thread, u64 store) -------------
__global__ void k_f32_to_bf16_v4(const float* __restrict__ src, bf16* __restrict__ dst,
                                 long long n4) {
  long long i = (long long)blockIdx.x * blockDim.x + threadIdx.x;
  if (i >= n4) return;
  float4 f = ((const float4*)src)[i];
  unsigned long long p = (unsigned long long)f2bf_u16(f.x)
                       | ((unsigned long long)f2bf_u16(f.y) << 16)
                       | ((unsigned long long)f2bf_u16(f.z) << 32)
                       | ((unsigned long long)f2bf_u16(f.w) << 48);
  ((unsigned long long*)dst)[i] = p;
}

// ---- C init: broadcast bias across rows ------------------------------------
__global__ void k_bias_rows(float* __restrict__ C, const float* __restrict__ bias,
                            int total, int ncols) {
  int i = blockIdx.x * blockDim.x + threadIdx.x;
  if (i < total) C[i] = bias[i % ncols];
}

__global__ void k_h_init(const float* __restrict__ h0, float* __restrict__ hpad) {
  int i = blockIdx.x * blockDim.x + threadIdx.x;
  if (i < 16 * HID) hpad[i] = (i < 8 * HID) ? h0[i] : 0.0f;
}

// ---- bf16 WMMA GEMM: C[m][n] += sum_k A[m][k] * W[n][k], K = HID fixed -----
// Single-wave blocks (32 threads, EXEC all ones). A: (rows x HID) bf16
// row-major, W: (N3 x HID) bf16 row-major. K-split across blockIdx.y, atomic
// f32 accumulate into bias-preinitialized C (stride N3). MT m-tiles per wave
// reuse each B fragment; all MT/NT row strides are compile-time constants
// (mt*16*HID*2 bytes < 2^23) folded into instruction offsets from 2 pointers.
// Double-buffered: fragments for K+32 are fetched before WMMAs on K issue.
// Fragment layouts used here (A/B share one contiguous k order):
//   A 16x32: lane(l16,half): row m=l16, K [k+half*16 .. +15] (one 32B run)
//   B 32x16: lane(l16,half): col n=l16, K [k+half*16 .. +15] (one 32B run)
//   C 16x16: vgpr r -> M = r + half*8, N = l16
// NTW=true  -> W loads non-temporal (one-shot stream, keep L2 for A/C).
// NTW=false -> W loads regular + global_prefetch_b8 at distance (re-read set).
template <int MT, int NT, bool NTW>
__global__ void k_gemm_bf16(const bf16* __restrict__ A, const bf16* __restrict__ W,
                            float* __restrict__ C, int kchunk) {
  constexpr int K = HID;
  const int lane = threadIdx.x;
  const int l16 = lane & 15, half = lane >> 4;
  const int nbase0 = blockIdx.x * (NT * 16);
  const int k0 = blockIdx.y * kchunk;      // kchunk % 64 == 0 (caller guarantees)
  const int mbase0 = blockIdx.z * (MT * 16);

  v8f acc[MT][NT] = {};

  // Two base pointers; per-tile row strides are compile-time immediates.
  const bf16* abase = A + (size_t)(mbase0 + l16) * K;
  const bf16* wbase = W + (size_t)(nbase0 + l16) * K;

  auto lda = [&](int kk, v16bf (&af)[MT]) {
#pragma unroll
    for (int mt = 0; mt < MT; ++mt) {
      af[mt] = *(const v16bf*)(abase + mt * (16 * K) + kk + half * 16);
    }
  };
  auto ldb = [&](int kk, v16bf (&bfr)[NT]) {
#pragma unroll
    for (int nt = 0; nt < NT; ++nt) {
      const v16bf* p = (const v16bf*)(wbase + nt * (16 * K) + kk + half * 16);
      if (NTW) {
        bfr[nt] = __builtin_nontemporal_load(p);
      } else {
        __builtin_prefetch(wbase + nt * (16 * K) + kk + 512, 0, 1);
        bfr[nt] = *p;
      }
    }
  };
  auto mm = [&](v16bf (&af)[MT], v16bf (&bfr)[NT]) {
#pragma unroll
    for (int nt = 0; nt < NT; ++nt)
#pragma unroll
      for (int mt = 0; mt < MT; ++mt)
        acc[mt][nt] = __builtin_amdgcn_wmma_f32_16x16x32_bf16(
            false, af[mt], false, bfr[nt], (short)0, acc[mt][nt], false, false);
  };

  v16bf a0[MT], a1[MT];
  v16bf b0[NT], b1[NT];
  const int kend = k0 + kchunk;
  lda(k0, a0); ldb(k0, b0);
  for (int k = k0; k < kend; k += 64) {
    lda(k + 32, a1); ldb(k + 32, b1);      // next fragments in flight
    mm(a0, b0);                            // compute on resident fragments
    if (k + 64 < kend) { lda(k + 64, a0); ldb(k + 64, b0); }
    mm(a1, b1);
  }

#pragma unroll
  for (int mt = 0; mt < MT; ++mt)
#pragma unroll
    for (int nt = 0; nt < NT; ++nt)
#pragma unroll
      for (int r = 0; r < 8; ++r) {
        int m = mbase0 + mt * 16 + r + half * 8;
        int n = nbase0 + nt * 16 + l16;
        atomicAdd(&C[(size_t)m * N3 + n], acc[mt][nt][r]);
      }
}

// ---- GRU gate fusion -------------------------------------------------------
__global__ void k_gru_update(const float* __restrict__ Gi_t, const float* __restrict__ gh,
                             float* __restrict__ hpad) {
  int i = blockIdx.x * blockDim.x + threadIdx.x;
  if (i >= 8 * HID) return;
  int b = i / HID, j = i % HID;
  const float* gi = Gi_t + (size_t)b * N3;
  const float* gr = gh + (size_t)b * N3;
  float i_r = gi[j], i_z = gi[HID + j], i_n = gi[2 * HID + j];
  float h_r = gr[j], h_z = gr[HID + j], h_n = gr[2 * HID + j];
  float r = 1.0f / (1.0f + expf(-(i_r + h_r)));
  float z = 1.0f / (1.0f + expf(-(i_z + h_z)));
  float n = tanhf(i_n + r * h_n);
  hpad[i] = (1.0f - z) * n + z * hpad[i];
}

// ---- final fc + labels -----------------------------------------------------
__global__ void k_fc2(const float* __restrict__ hpad, const float* __restrict__ w,
                      const float* __restrict__ bias, const long long* __restrict__ labels,
                      float* __restrict__ out, int out_n) {
  int i = threadIdx.x;
  if (i < 80) {
    int b = i / 10, c = i % 10;
    float s = bias[c];
    const float* hr = hpad + (size_t)b * HID;
    const float* wr = w + (size_t)c * HID;
    for (int k = 0; k < HID; ++k) s += hr[k] * wr[k];
    out[i] = s;
  } else if (i >= 80 && i < 88) {
    if (i < out_n) out[i] = (float)labels[i - 80];
  }
}

extern "C" void kernel_launch(void* const* d_in, const int* in_sizes, int n_in,
                              void* d_out, int out_size, void* d_ws, size_t ws_size,
                              hipStream_t stream) {
  const float*     x      = (const float*)d_in[0];
  const long long* labels = (const long long*)d_in[1];
  const float*     h0     = (const float*)d_in[2];
  const float*     c1w    = (const float*)d_in[3];
  const float*     c1b    = (const float*)d_in[4];
  const float*     c2w    = (const float*)d_in[5];
  const float*     c2b    = (const float*)d_in[6];
  const float*     w_ih   = (const float*)d_in[7];
  const float*     w_hh   = (const float*)d_in[8];
  const float*     b_ih   = (const float*)d_in[9];
  const float*     b_hh   = (const float*)d_in[10];
  const float*     fc2w   = (const float*)d_in[11];
  const float*     fc2b   = (const float*)d_in[12];
  (void)in_sizes; (void)n_in; (void)ws_size;

  char* wsb = (char*)d_ws;
  size_t off = 0;
  auto carve = [&](size_t bytes) -> void* {
    void* p = wsb + off;
    off += (bytes + 255) & ~(size_t)255;
    return p;
  };
  const size_t WN = (size_t)N3 * HID;                 // 118,013,952 weights each
  bf16*  wih_bf   = (bf16*) carve(WN * 2);            // 236 MB
  bf16*  whh_bf   = (bf16*) carve(WN * 2);            // 236 MB
  float* c1out    = (float*)carve((size_t)NBAT * 8 * 84 * 41 * 4);
  float* featsA   = (float*)carve((size_t)NBAT * HID * 4);
  bf16*  feats_bf = (bf16*) carve((size_t)NBAT * HID * 2);
  float* Gi       = (float*)carve((size_t)NBAT * N3 * 4);
  float* gh       = (float*)carve((size_t)16 * N3 * 4);
  float* hpad     = (float*)carve((size_t)16 * HID * 4);
  bf16*  h_bf     = (bf16*) carve((size_t)16 * HID * 2);

  // 1) CNN front-end
  {
    long long n = (long long)NBAT * 8 * 84 * 41;
    k_conv1_pool<<<(unsigned)((n + 255) / 256), 256, 0, stream>>>(x, c1w, c1b, c1out);
  }
  {
    long long n = (long long)NBAT * 16 * 28 * 14;
    k_conv2_pool<<<(unsigned)((n + 255) / 256), 256, 0, stream>>>(c1out, c2w, c2b, featsA);
  }

  // 2) bf16 packing (weights once per launch; feats tiny)
  {
    long long n4 = (long long)(WN / 4);
    unsigned g = (unsigned)((n4 + 255) / 256);
    k_f32_to_bf16_v4<<<g, 256, 0, stream>>>(w_ih, wih_bf, n4);
    k_f32_to_bf16_v4<<<g, 256, 0, stream>>>(w_hh, whh_bf, n4);
    long long f4 = (long long)NBAT * HID / 4;
    k_f32_to_bf16_v4<<<(unsigned)((f4 + 255) / 256), 256, 0, stream>>>(featsA, feats_bf, f4);
  }

  // 3) Gi = feats @ w_ih^T + b_ih for ALL 20 timesteps at once (M=160)
  {
    int total = NBAT * N3;
    k_bias_rows<<<(total + 255) / 256, 256, 0, stream>>>(Gi, b_ih, total, N3);
    dim3 g(N3 / 16, 2, 2);                 // 1176 n-tiles x ksplit2 x 2 m-groups
    k_gemm_bf16<5, 1, true><<<g, 32, 0, stream>>>(feats_bf, wih_bf, Gi, HID / 2);
  }

  // 4) sequential GRU: gh = h @ w_hh^T + b_hh each step (M padded 8->16)
  k_h_init<<<(16 * HID + 255) / 256, 256, 0, stream>>>(h0, hpad);
  for (int t = 0; t < NWIN; ++t) {
    int total = 16 * N3;
    k_bias_rows<<<(total + 255) / 256, 256, 0, stream>>>(gh, b_hh, total, N3);
    long long h4 = 16LL * HID / 4;
    k_f32_to_bf16_v4<<<(unsigned)((h4 + 255) / 256), 256, 0, stream>>>(hpad, h_bf, h4);
    dim3 g(N3 / 64, 7, 1);                 // 294 n-groups(NT=4) x ksplit7
    k_gemm_bf16<1, 4, false><<<g, 32, 0, stream>>>(h_bf, whh_bf, gh, HID / 7);
    k_gru_update<<<(8 * HID + 255) / 256, 256, 0, stream>>>(
        Gi + (size_t)t * 8 * N3, gh, hpad);
  }

  // 5) logits + labels
  k_fc2<<<1, 128, 0, stream>>>(hpad, fc2w, fc2b, labels, (float*)d_out, out_size);
}
